// ReflexPolicy_73065983640288
// MI455X (gfx1250) — compile-verified, matrix operand order checked by reading, not run on hardware
//
#include <hip/hip_runtime.h>
#include <hip/hip_bf16.h>

typedef __attribute__((ext_vector_type(16))) _Float16 v16h;
typedef __attribute__((ext_vector_type(8)))  _Float16 h8;
typedef __attribute__((ext_vector_type(8)))  float    v8f;

#define B_TOTAL 8192
#define OBS 64
#define ACT 32
#define LAT 25
#define RH  5
#define SUPH 64
#define OA  (OBS*ACT)   // 2048

// ---------------------------------------------------------------------------
// Operand loaders. Row-major storage with `strideH` halfs per row.
// 16-bit A-matrix 16x32 layout (ISA 7.12.2): lanes 0-15 hold row M=lane,
// K=0..7 (VGPR0-3) and K=16..23 (VGPR4-7); lanes 16-31 hold the same row,
// K=8..15 and K=24..31.  => two 16-byte chunks per lane.
// B operand assumed to be the A-layout of B^T (store W transposed, rows = N).
// ---------------------------------------------------------------------------
__device__ __forceinline__ v16h load_tile_k32(const _Float16* __restrict__ base,
                                              int lane, int kc, int strideH) {
    const int row   = lane & 15;
    const int hiOff = (lane & 16) ? 8 : 0;              // elements
    const _Float16* r = base + row * strideH + kc + hiOff;
    h8 lo = *(const h8*)(r);
    h8 hi = *(const h8*)(r + 16);
    return __builtin_shufflevector(lo, hi, 0,1,2,3,4,5,6,7,8,9,10,11,12,13,14,15);
}

__device__ __forceinline__ float red16_max(float v) {
    v = fmaxf(v, __shfl_xor(v, 1));
    v = fmaxf(v, __shfl_xor(v, 2));
    v = fmaxf(v, __shfl_xor(v, 4));
    v = fmaxf(v, __shfl_xor(v, 8));
    return v;
}
__device__ __forceinline__ float red16_sum(float v) {
    v += __shfl_xor(v, 1);
    v += __shfl_xor(v, 2);
    v += __shfl_xor(v, 4);
    v += __shfl_xor(v, 8);
    return v;
}

// ---------------------------------------------------------------------------
// Kernel 0: act_matrix (LAT,OBS,ACT) f32  ->  AMt (OA rows x 32 halfs), K padded
// ---------------------------------------------------------------------------
__global__ __launch_bounds__(256) void prep_amt_kernel(
        const float* __restrict__ am, _Float16* __restrict__ amt) {
    int idx = blockIdx.x * 256 + threadIdx.x;           // 0 .. OA*32-1
    if (idx >= OA * 32) return;
    int oa = idx >> 5;
    int l  = idx & 31;
    amt[oa * 32 + l] = (l < LAT) ? (_Float16)am[l * OA + oa] : (_Float16)0.0f;
}

// ---------------------------------------------------------------------------
// Kernel 1: supervisor MLP -> latent (B x 32) f16, cols 25..31 zeroed.
// 8 waves / block, 16 batch rows / wave, 64 blocks.
// ---------------------------------------------------------------------------
__global__ __launch_bounds__(256) void supervisor_kernel(
        const float* __restrict__ obs,
        const float* __restrict__ sw1, const float* __restrict__ sb1,
        const float* __restrict__ sw2, const float* __restrict__ sb2,
        const float* __restrict__ sw3, const float* __restrict__ sb3,
        _Float16* __restrict__ latent) {
    __shared__ __align__(16) _Float16 w1T[SUPH * SUPH];   // [n][k]
    __shared__ __align__(16) _Float16 w2T[SUPH * SUPH];
    __shared__ __align__(16) _Float16 w3T[32 * SUPH];     // rows 25..31 zero
    __shared__ __align__(16) _Float16 actA[8][16 * SUPH];
    __shared__ __align__(16) _Float16 actB[8][16 * SUPH];

    const int tid  = threadIdx.x;
    const int lane = tid & 31;
    const int wave = tid >> 5;
    const int rowBase = blockIdx.x * 128 + wave * 16;
    const int hi8  = (lane & 16) ? 8 : 0;
    const int nl   = lane & 15;

    // stage weights (transposed, f16)
    for (int idx = tid; idx < SUPH * SUPH; idx += 256) {
        int k = idx >> 6, n = idx & 63;
        w1T[n * SUPH + k] = (_Float16)sw1[idx];
        w2T[n * SUPH + k] = (_Float16)sw2[idx];
    }
    for (int idx = tid; idx < 32 * SUPH; idx += 256) {
        int k = idx >> 5, n = idx & 31;
        w3T[n * SUPH + k] = (n < LAT) ? (_Float16)sw3[k * LAT + n] : (_Float16)0.0f;
    }
    // stage obs block (16 rows x 64 cols, contiguous) into actA[wave]
    for (int idx = lane; idx < 16 * OBS; idx += 32)
        actA[wave][idx] = (_Float16)obs[rowBase * OBS + idx];
    __syncthreads();

    // ---- layer 1: actA -> actB, ReLU ----
    #pragma unroll
    for (int t = 0; t < 4; ++t) {
        v8f c = {};
        #pragma unroll
        for (int kc = 0; kc < SUPH; kc += 32) {
            v16h a = load_tile_k32(actA[wave], lane, kc, SUPH);
            v16h b = load_tile_k32(w1T + t * 16 * SUPH, lane, kc, SUPH);
            c = __builtin_amdgcn_wmma_f32_16x16x32_f16(false, a, false, b,
                                                       (short)0, c, false, false);
        }
        int n = t * 16 + nl;
        float bv = sb1[n];
        #pragma unroll
        for (int i = 0; i < 8; ++i)
            actB[wave][(i + hi8) * SUPH + n] = (_Float16)fmaxf(c[i] + bv, 0.0f);
    }
    __syncthreads();

    // ---- layer 2: actB -> actA, ReLU ----
    #pragma unroll
    for (int t = 0; t < 4; ++t) {
        v8f c = {};
        #pragma unroll
        for (int kc = 0; kc < SUPH; kc += 32) {
            v16h a = load_tile_k32(actB[wave], lane, kc, SUPH);
            v16h b = load_tile_k32(w2T + t * 16 * SUPH, lane, kc, SUPH);
            c = __builtin_amdgcn_wmma_f32_16x16x32_f16(false, a, false, b,
                                                       (short)0, c, false, false);
        }
        int n = t * 16 + nl;
        float bv = sb2[n];
        #pragma unroll
        for (int i = 0; i < 8; ++i)
            actA[wave][(i + hi8) * SUPH + n] = (_Float16)fmaxf(c[i] + bv, 0.0f);
    }
    __syncthreads();

    // ---- layer 3: actA -> latent (global), sigmoid, zero-pad cols >= LAT ----
    #pragma unroll
    for (int t = 0; t < 2; ++t) {
        v8f c = {};
        #pragma unroll
        for (int kc = 0; kc < SUPH; kc += 32) {
            v16h a = load_tile_k32(actA[wave], lane, kc, SUPH);
            v16h b = load_tile_k32(w3T + t * 16 * SUPH, lane, kc, SUPH);
            c = __builtin_amdgcn_wmma_f32_16x16x32_f16(false, a, false, b,
                                                       (short)0, c, false, false);
        }
        int n = t * 16 + nl;
        float bv = (n < LAT) ? sb3[n] : 0.0f;
        #pragma unroll
        for (int i = 0; i < 8; ++i) {
            float sg = 1.0f / (1.0f + __expf(-(c[i] + bv)));
            _Float16 hv = (n < LAT) ? (_Float16)sg : (_Float16)0.0f;
            latent[(rowBase + i + hi8) * 32 + n] = hv;
        }
    }
}

// ---------------------------------------------------------------------------
// Kernel 2: gating GEMM (WMMA, K=32) + row-softmax + reflex MLPs + reduce.
// Software-pipelined: B operands / reflex params for iteration o+1 are issued
// right after the WMMAs of iteration o, so their latency hides under the
// ~240 VALU ops of the reflex MLP + softmax (which co-execute with WMMA).
// ---------------------------------------------------------------------------
__device__ __forceinline__ v16h load_b_tile(const _Float16* __restrict__ amt,
                                            int o, int t, int a0, int hiOff) {
    const _Float16* r = amt + (32 * o + t * 16 + a0) * 32 + hiOff;
    h8 lo = *(const h8*)(r);
    h8 hi = *(const h8*)(r + 16);
    return __builtin_shufflevector(lo, hi, 0,1,2,3,4,5,6,7,8,9,10,11,12,13,14,15);
}

__global__ __launch_bounds__(256) void reflex_kernel(
        const float* __restrict__ obs,
        const _Float16* __restrict__ latent,
        const _Float16* __restrict__ amt,
        const float* __restrict__ rw1, const float* __restrict__ rb1,
        const float* __restrict__ rw2, const float* __restrict__ rb2,
        float* __restrict__ out) {
    __shared__ float obs_s[8][16 * OBS];

    const int tid  = threadIdx.x;
    const int lane = tid & 31;
    const int wave = tid >> 5;
    const int rowBase = blockIdx.x * 128 + wave * 16;
    const int hi8  = (lane & 16) ? 8 : 0;
    const int a0   = lane & 15;          // ACT column of tile 0; tile 1 is a0+16

    for (int idx = lane; idx < 16 * OBS; idx += 32)
        obs_s[wave][idx] = obs[rowBase * OBS + idx];
    __syncthreads();

    // A operand: latent tile (16 rows x K=32), loaded once, reused for all o.
    const int arow   = rowBase + (lane & 15);
    const int ahiOff = (lane & 16) ? 8 : 0;
    h8 alo = *(const h8*)(latent + arow * 32 + ahiOff);
    h8 ahi = *(const h8*)(latent + arow * 32 + 16 + ahiOff);
    v16h amat = __builtin_shufflevector(alo, ahi,
                                        0,1,2,3,4,5,6,7,8,9,10,11,12,13,14,15);

    float acc0[8], acc1[8];
    #pragma unroll
    for (int i = 0; i < 8; ++i) { acc0[i] = 0.0f; acc1[i] = 0.0f; }

    // ---- pipeline prologue: operands for o = 0 ----
    v16h b0 = load_b_tile(amt, 0, 0, a0, ahiOff);
    v16h b1 = load_b_tile(amt, 0, 1, a0, ahiOff);
    float w1a[RH], b1a[RH], w2a[RH], w1b[RH], b1b[RH], w2b[RH], c2a, c2b;
    {
        const int pa = a0 * RH, pb = (a0 + 16) * RH;
        #pragma unroll
        for (int h = 0; h < RH; ++h) {
            w1a[h] = rw1[pa + h]; b1a[h] = rb1[pa + h]; w2a[h] = rw2[pa + h];
            w1b[h] = rw1[pb + h]; b1b[h] = rb1[pb + h]; w2b[h] = rw2[pb + h];
        }
        c2a = rb2[a0]; c2b = rb2[a0 + 16];
    }

    for (int o = 0; o < OBS; ++o) {
        v8f z = {};
        v8f c0 = __builtin_amdgcn_wmma_f32_16x16x32_f16(false, amat, false, b0,
                                                        (short)0, z, false, false);
        v8f c1 = __builtin_amdgcn_wmma_f32_16x16x32_f16(false, amat, false, b1,
                                                        (short)0, z, false, false);

        // ---- issue next iteration's loads now (latency hidden by VALU below)
        const int on = (o + 1 < OBS) ? o + 1 : o;       // uniform, branch-free
        v16h nb0 = load_b_tile(amt, on, 0, a0, ahiOff);
        v16h nb1 = load_b_tile(amt, on, 1, a0, ahiOff);
        float nw1a[RH], nb1a[RH], nw2a[RH], nw1b[RH], nb1b[RH], nw2b[RH];
        const int pa = (on * ACT + a0) * RH;
        const int pb = (on * ACT + a0 + 16) * RH;
        #pragma unroll
        for (int h = 0; h < RH; ++h) {
            nw1a[h] = rw1[pa + h]; nb1a[h] = rb1[pa + h]; nw2a[h] = rw2[pa + h];
            nw1b[h] = rw1[pb + h]; nb1b[h] = rb1[pb + h]; nw2b[h] = rw2[pb + h];
        }
        const float nc2a = rb2[on * ACT + a0];
        const float nc2b = rb2[on * ACT + a0 + 16];
        // prefetch the AMt stream two o-groups ahead (global_prefetch_b8)
        if (o + 2 < OBS)
            __builtin_prefetch(amt + (size_t)(32 * (o + 2) + a0) * 32, 0, 0);

        // ---- reflex MLPs + fused row-softmax + weighted accumulate ----
        #pragma unroll
        for (int i = 0; i < 8; ++i) {
            float x  = obs_s[wave][(i + hi8) * OBS + o];
            float r0 = c2a, r1 = c2b;
            #pragma unroll
            for (int h = 0; h < RH; ++h) {
                r0 += fmaxf(fmaf(x, w1a[h], b1a[h]), 0.0f) * w2a[h];
                r1 += fmaxf(fmaf(x, w1b[h], b1b[h]), 0.0f) * w2b[h];
            }
            // row-wise softmax over 32 ACT logits (16-lane-half reduction)
            float m  = red16_max(fmaxf(c0[i], c1[i]));
            float e0 = __expf(c0[i] - m);
            float e1 = __expf(c1[i] - m);
            float inv = 1.0f / red16_sum(e0 + e1);
            acc0[i] += (e0 * inv) * r0;
            acc1[i] += (e1 * inv) * r1;
        }

        // ---- rotate pipeline registers ----
        b0 = nb0; b1 = nb1;
        #pragma unroll
        for (int h = 0; h < RH; ++h) {
            w1a[h] = nw1a[h]; b1a[h] = nb1a[h]; w2a[h] = nw2a[h];
            w1b[h] = nw1b[h]; b1b[h] = nb1b[h]; w2b[h] = nw2b[h];
        }
        c2a = nc2a; c2b = nc2b;
    }

    #pragma unroll
    for (int i = 0; i < 8; ++i) {
        int row = rowBase + i + hi8;
        out[row * ACT + a0]      = acc0[i];
        out[row * ACT + a0 + 16] = acc1[i];
    }
}

// ---------------------------------------------------------------------------
extern "C" void kernel_launch(void* const* d_in, const int* in_sizes, int n_in,
                              void* d_out, int out_size, void* d_ws, size_t ws_size,
                              hipStream_t stream) {
    const float* obs = (const float*)d_in[0];
    const float* sw1 = (const float*)d_in[1];
    const float* sb1 = (const float*)d_in[2];
    const float* sw2 = (const float*)d_in[3];
    const float* sb2 = (const float*)d_in[4];
    const float* sw3 = (const float*)d_in[5];
    const float* sb3 = (const float*)d_in[6];
    const float* rw1 = (const float*)d_in[7];
    const float* rb1 = (const float*)d_in[8];
    const float* rw2 = (const float*)d_in[9];
    const float* rb2 = (const float*)d_in[10];
    const float* am  = (const float*)d_in[11];

    _Float16* latent = (_Float16*)d_ws;                          // B*32 halfs
    _Float16* amt    = (_Float16*)((char*)d_ws +
                                   (size_t)B_TOTAL * 32 * sizeof(_Float16));

    prep_amt_kernel<<<(OA * 32 + 255) / 256, 256, 0, stream>>>(am, amt);
    supervisor_kernel<<<B_TOTAL / 128, 256, 0, stream>>>(
        obs, sw1, sb1, sw2, sb2, sw3, sb3, latent);
    reflex_kernel<<<B_TOTAL / 128, 256, 0, stream>>>(
        obs, latent, amt, rw1, rb1, rw2, rb2, (float*)d_out);
}
